// SparseBSRLinear_64639257805130
// MI455X (gfx1250) — matrix operand order repaired and checked by reading, not compiled, not run on hardware
//
#include <hip/hip_runtime.h>

// Block-sparse BSR linear: y = x @ W.T + bias on gfx1250.
// bf16x3 WMMA emulation of fp32 GEMM + precomputed hi/lo planes + async LDS staging.
#define OUT_F  4096
#define IN_F   4096
#define BS     64
#define KBLK   32      // nonzero blocks per block-row
#define NROWB  64
#define NNZB   (NROWB * KBLK)
#define BATCH  8192
#define BM     128     // batch rows per workgroup
#define XS     72      // padded LDS row stride (bf16 elems): 144B rows, 16B aligned, bank-spread

#define AS1 __attribute__((address_space(1)))
#define AS3 __attribute__((address_space(3)))

typedef __attribute__((ext_vector_type(16))) __bf16 v16bf;
typedef __attribute__((ext_vector_type(8)))  float  v8f;
typedef int b128_t __attribute__((vector_size(16)));   // matches builtin's int4 pointee

union Frag {
    v16bf v;
    unsigned int u[8];
};

__device__ __forceinline__ unsigned short f2bf_rne(float f) {
    unsigned int u = __float_as_uint(f);
    u += 0x7FFFu + ((u >> 16) & 1u);      // round-to-nearest-even
    return (unsigned short)(u >> 16);
}
__device__ __forceinline__ float bf2f(unsigned short h) {
    return __uint_as_float(((unsigned int)h) << 16);
}
__device__ __forceinline__ void split_bf16(float f, unsigned short& hi, unsigned short& lo) {
    hi = f2bf_rne(f);
    lo = f2bf_rne(f - bf2f(hi));
}

// ---------------- async global->LDS copy (CDNA5) with fallback ----------------
#if defined(__HIP_DEVICE_COMPILE__) && __has_builtin(__builtin_amdgcn_global_load_async_to_lds_b128)
#define HAVE_ASYNC_LDS 1
#else
#define HAVE_ASYNC_LDS 0
#endif

__device__ __forceinline__ void copy16(void* ldst, const void* gsrc) {
#if HAVE_ASYNC_LDS
    __builtin_amdgcn_global_load_async_to_lds_b128(
        (AS1 b128_t*)(uintptr_t)gsrc,
        (AS3 b128_t*)(unsigned int)(uintptr_t)ldst, 0, 0);
#else
    *(uint4*)ldst = *(const uint4*)gsrc;
#endif
}

__device__ __forceinline__ void wait_async0() {
#if defined(__HIP_DEVICE_COMPILE__)
#if __has_builtin(__builtin_amdgcn_s_wait_asynccnt)
    __builtin_amdgcn_s_wait_asynccnt(0);
#else
    asm volatile("s_wait_asynccnt 0x0" ::: "memory");
#endif
#endif
}

// ---------------- pass 1/2: fp32 -> bf16 hi/lo plane split ----------------
__global__ __launch_bounds__(256)
void split_f32_bf16x2(const float* __restrict__ in,
                      unsigned short* __restrict__ hi,
                      unsigned short* __restrict__ lo, int n4) {
    int i = blockIdx.x * 256 + threadIdx.x;
    if (i >= n4) return;
    float4 f = ((const float4*)in)[i];
    ushort4 hv, lv;
    split_bf16(f.x, hv.x, lv.x);
    split_bf16(f.y, hv.y, lv.y);
    split_bf16(f.z, hv.z, lv.z);
    split_bf16(f.w, hv.w, lv.w);
    ((ushort4*)hi)[i] = hv;
    ((ushort4*)lo)[i] = lv;
}

// ---------------- main GEMM: double-buffered async staging + bf16x3 WMMA ----------------
__global__ __launch_bounds__(256)
void SparseBSRLinear_wmma_async(const unsigned short* __restrict__ xhi_g,
                                const unsigned short* __restrict__ xlo_g,
                                const unsigned short* __restrict__ whi_g,
                                const unsigned short* __restrict__ wlo_g,
                                const float* __restrict__ bias,
                                const int*   __restrict__ col_indices,
                                float*       __restrict__ out) {
    __shared__ unsigned short sxh[2][BM * XS];
    __shared__ unsigned short sxl[2][BM * XS];
    __shared__ unsigned short swh[2][BS * XS];
    __shared__ unsigned short swl[2][BS * XS];

    const int tid  = threadIdx.x;
    const int wave = tid >> 5;
    const int lane = tid & 31;
    const int lr   = lane & 15;
    const int lhi  = lane >> 4;

    const int batchBase = blockIdx.x * BM;
    const int r0        = blockIdx.y;

    const v8f zero = {0.f, 0.f, 0.f, 0.f, 0.f, 0.f, 0.f, 0.f};
    v8f acc[4] = {zero, zero, zero, zero};

    const int arow = (wave << 4) + lr;

    // stage tile kb into buffer buf (12 x b128 copies per thread)
    auto stage = [&](int kb, int buf) {
        const int blk = r0 * KBLK + kb;
        const int c   = col_indices[blk];
        #pragma unroll
        for (int t = 0; t < 4; ++t) {               // x planes: 1024 b128 chunks each
            int idx = tid + t * 256;
            int row = idx >> 3, c8 = idx & 7;
            size_t go = (size_t)(batchBase + row) * IN_F + c * BS + (c8 << 3);
            int    lo = row * XS + (c8 << 3);
            copy16(&sxh[buf][lo], &xhi_g[go]);
            copy16(&sxl[buf][lo], &xlo_g[go]);
        }
        #pragma unroll
        for (int t = 0; t < 2; ++t) {               // w planes: 512 b128 chunks each
            int idx = tid + t * 256;
            int row = idx >> 3, c8 = idx & 7;
            size_t go = (size_t)blk * (BS * BS) + ((size_t)idx << 3);
            int    lo = row * XS + (c8 << 3);
            copy16(&swh[buf][lo], &whi_g[go]);
            copy16(&swl[buf][lo], &wlo_g[go]);
        }
    };

    stage(0, 0);
    wait_async0();
    __syncthreads();

    for (int kb = 0; kb < KBLK; ++kb) {
        const int buf = kb & 1;
        if (kb + 1 < KBLK) stage(kb + 1, buf ^ 1);   // async prefetch into other buffer

        #pragma unroll
        for (int kc = 0; kc < 2; ++kc) {
            // A frag (16x32): lanes 0-15 hold K {0..7,16..23}, lanes 16-31 K {8..15,24..31}
            Frag ahi, alo;
            #pragma unroll
            for (int v = 0; v < 8; ++v) {
                int k   = kc * 32 + ((v < 4) ? (v << 1) : ((v << 1) + 8)) + (lhi << 3);
                int off = arow * XS + k;
                ahi.u[v] = *(const unsigned int*)&sxh[buf][off];
                alo.u[v] = *(const unsigned int*)&sxl[buf][off];
            }
            #pragma unroll
            for (int nt = 0; nt < 4; ++nt) {
                // B frag (32x16): B[k][n] = Wblk[n][k]; lanes 0-15 K {0..7,16..23}, lanes 16-31 K {8..15,24..31}
                Frag bhi, blo;
                const int ncol = (nt << 4) + lr;
                #pragma unroll
                for (int v = 0; v < 8; ++v) {
                    int k   = kc * 32 + ((v < 4) ? (v << 1) : ((v << 1) + 8)) + (lhi << 3);
                    int off = ncol * XS + k;
                    bhi.u[v] = *(const unsigned int*)&swh[buf][off];
                    blo.u[v] = *(const unsigned int*)&swl[buf][off];
                }
                acc[nt] = __builtin_amdgcn_wmma_f32_16x16x32_bf16(
                    false, ahi.v, false, bhi.v, (short)0, acc[nt], false, false);
                acc[nt] = __builtin_amdgcn_wmma_f32_16x16x32_bf16(
                    false, ahi.v, false, blo.v, (short)0, acc[nt], false, false);
                acc[nt] = __builtin_amdgcn_wmma_f32_16x16x32_bf16(
                    false, alo.v, false, bhi.v, (short)0, acc[nt], false, false);
            }
        }
        wait_async0();      // prefetch landed
        __syncthreads();    // all waves done reading buf before it is re-staged
    }

    const int colBase = r0 * BS;
    #pragma unroll
    for (int nt = 0; nt < 4; ++nt) {
        const int cn = colBase + (nt << 4) + lr;
        const float bv = bias[cn];
        #pragma unroll
        for (int vr = 0; vr < 8; ++vr) {
            int row = batchBase + (wave << 4) + (lhi << 3) + vr;
            out[(size_t)row * OUT_F + cn] = acc[nt][vr] + bv;
        }
    }
}

// ---------------- fallback: fused split-in-kernel version (small workspace) ----------------
__global__ __launch_bounds__(256)
void SparseBSRLinear_wmma_fused(const float* __restrict__ x,
                                const float* __restrict__ values,
                                const float* __restrict__ bias,
                                const int*   __restrict__ col_indices,
                                float*       __restrict__ out) {
    __shared__ unsigned short xhi[BM * XS];
    __shared__ unsigned short xlo[BM * XS];
    __shared__ unsigned short whi[BS * XS];
    __shared__ unsigned short wlo[BS * XS];

    const int tid  = threadIdx.x;
    const int wave = tid >> 5;
    const int lane = tid & 31;
    const int lr   = lane & 15;
    const int lhi  = lane >> 4;

    const int batchBase = blockIdx.x * BM;
    const int r0        = blockIdx.y;

    const v8f zero = {0.f, 0.f, 0.f, 0.f, 0.f, 0.f, 0.f, 0.f};
    v8f acc[4] = {zero, zero, zero, zero};
    const int arow = (wave << 4) + lr;

    for (int kb = 0; kb < KBLK; ++kb) {
        const int blk = r0 * KBLK + kb;
        const int c   = col_indices[blk];
        __syncthreads();
        #pragma unroll
        for (int t = 0; t < 8; ++t) {
            int idx = tid + t * 256;
            int row = idx >> 4, c4 = idx & 15;
            const float4 f =
                ((const float4*)(x + (size_t)(batchBase + row) * IN_F + c * BS))[c4];
            unsigned short h0, h1, h2, h3, l0, l1, l2, l3;
            split_bf16(f.x, h0, l0); split_bf16(f.y, h1, l1);
            split_bf16(f.z, h2, l2); split_bf16(f.w, h3, l3);
            int off = row * XS + (c4 << 2);
            *(unsigned int*)&xhi[off]     = (unsigned int)h0 | ((unsigned int)h1 << 16);
            *(unsigned int*)&xhi[off + 2] = (unsigned int)h2 | ((unsigned int)h3 << 16);
            *(unsigned int*)&xlo[off]     = (unsigned int)l0 | ((unsigned int)l1 << 16);
            *(unsigned int*)&xlo[off + 2] = (unsigned int)l2 | ((unsigned int)l3 << 16);
        }
        {
            const float4* wsrc = (const float4*)(values + (size_t)blk * BS * BS);
            #pragma unroll
            for (int t = 0; t < 4; ++t) {
                int idx = tid + t * 256;
                int row = idx >> 4, c4 = idx & 15;
                const float4 f = wsrc[idx];
                unsigned short h0, h1, h2, h3, l0, l1, l2, l3;
                split_bf16(f.x, h0, l0); split_bf16(f.y, h1, l1);
                split_bf16(f.z, h2, l2); split_bf16(f.w, h3, l3);
                int off = row * XS + (c4 << 2);
                *(unsigned int*)&whi[off]     = (unsigned int)h0 | ((unsigned int)h1 << 16);
                *(unsigned int*)&whi[off + 2] = (unsigned int)h2 | ((unsigned int)h3 << 16);
                *(unsigned int*)&wlo[off]     = (unsigned int)l0 | ((unsigned int)l1 << 16);
                *(unsigned int*)&wlo[off + 2] = (unsigned int)l2 | ((unsigned int)l3 << 16);
            }
        }
        __syncthreads();
        #pragma unroll
        for (int kc = 0; kc < 2; ++kc) {
            Frag ahi, alo;
            #pragma unroll
            for (int v = 0; v < 8; ++v) {
                int k   = kc * 32 + ((v < 4) ? (v << 1) : ((v << 1) + 8)) + (lhi << 3);
                int off = arow * XS + k;
                ahi.u[v] = *(const unsigned int*)&xhi[off];
                alo.u[v] = *(const unsigned int*)&xlo[off];
            }
            #pragma unroll
            for (int nt = 0; nt < 4; ++nt) {
                Frag bhi, blo;
                const int ncol = (nt << 4) + lr;
                #pragma unroll
                for (int v = 0; v < 8; ++v) {
                    int k   = kc * 32 + ((v < 4) ? (v << 1) : ((v << 1) + 8)) + (lhi << 3);
                    int off = ncol * XS + k;
                    bhi.u[v] = *(const unsigned int*)&whi[off];
                    blo.u[v] = *(const unsigned int*)&wlo[off];
                }
                acc[nt] = __builtin_amdgcn_wmma_f32_16x16x32_bf16(
                    false, ahi.v, false, bhi.v, (short)0, acc[nt], false, false);
                acc[nt] = __builtin_amdgcn_wmma_f32_16x16x32_bf16(
                    false, ahi.v, false, blo.v, (short)0, acc[nt], false, false);
                acc[nt] = __builtin_amdgcn_wmma_f32_16x16x32_bf16(
                    false, alo.v, false, bhi.v, (short)0, acc[nt], false, false);
            }
        }
    }

    const int colBase = r0 * BS;
    #pragma unroll
    for (int nt = 0; nt < 4; ++nt) {
        const int cn = colBase + (nt << 4) + lr;
        const float bv = bias[cn];
        #pragma unroll
        for (int vr = 0; vr < 8; ++vr) {
            int row = batchBase + (wave << 4) + (lhi << 3) + vr;
            out[(size_t)row * OUT_F + cn] = acc[nt][vr] + bv;
        }
    }
}

extern "C" void kernel_launch(void* const* d_in, const int* in_sizes, int n_in,
                              void* d_out, int out_size, void* d_ws, size_t ws_size,
                              hipStream_t stream) {
    const float* x           = (const float*)d_in[0];
    const float* values      = (const float*)d_in[1];
    const float* bias        = (const float*)d_in[2];
    // d_in[3] = crow_indices: uniform (r*K), not needed
    const int*   col_indices = (const int*)d_in[4];
    float*       out         = (float*)d_out;

    const size_t X_ELEMS = (size_t)BATCH * IN_F;       // 33.5M
    const size_t W_ELEMS = (size_t)NNZB * BS * BS;     // 8.4M
    const size_t OFF_XHI = 0;
    const size_t OFF_XLO = OFF_XHI + X_ELEMS * 2;
    const size_t OFF_WHI = OFF_XLO + X_ELEMS * 2;
    const size_t OFF_WLO = OFF_WHI + W_ELEMS * 2;
    const size_t NEED    = OFF_WLO + W_ELEMS * 2;      // 160 MiB

    dim3 grid(BATCH / BM, NROWB);
    if (ws_size >= NEED) {
        unsigned short* xhi = (unsigned short*)((char*)d_ws + OFF_XHI);
        unsigned short* xlo = (unsigned short*)((char*)d_ws + OFF_XLO);
        unsigned short* whi = (unsigned short*)((char*)d_ws + OFF_WHI);
        unsigned short* wlo = (unsigned short*)((char*)d_ws + OFF_WLO);

        int xn4 = (int)(X_ELEMS / 4);
        int wn4 = (int)(W_ELEMS / 4);
        split_f32_bf16x2<<<xn4 / 256, 256, 0, stream>>>(x, xhi, xlo, xn4);
        split_f32_bf16x2<<<wn4 / 256, 256, 0, stream>>>(values, whi, wlo, wn4);
        SparseBSRLinear_wmma_async<<<grid, 256, 0, stream>>>(
            xhi, xlo, whi, wlo, bias, col_indices, out);
    } else {
        SparseBSRLinear_wmma_fused<<<grid, 256, 0, stream>>>(
            x, values, bias, col_indices, out);
    }
}
